// Sequence_7121055777315
// MI455X (gfx1250) — matrix operand order repaired and checked by reading, not compiled, not run on hardware
//
#include <hip/hip_runtime.h>
#include <math.h>

typedef __attribute__((ext_vector_type(16))) __bf16 v16bf;
typedef __attribute__((ext_vector_type(8)))  __bf16 v8bf;
typedef __attribute__((ext_vector_type(8)))  float  v8f;
typedef __attribute__((ext_vector_type(4)))  unsigned int uint32x4;

#define HDIM     256
#define BT       32        // batch rows per workgroup
#define HST      264       // padded h-buffer row stride (bf16 elems) to spread LDS banks
#define NTHREADS 256

// ---- LDS layout (byte offsets), carved from one dynamic allocation ----
#define WB_OFF     0
#define WB_BYTES   (16*8*32*16*2)            // 131072: B-fragments, [coltile][kchunk][lane][16 bf16]
#define HBUF_OFF   (WB_OFF + WB_BYTES)       // 131072
#define HBUF_BYTES (2*BT*HST*2)              // 33792: double-buffered h (bf16)
#define BIAS_OFF   (HBUF_OFF + HBUF_BYTES)   // 164864: b_ih + b_hh (f32[256])
#define WIH_OFF    (BIAS_OFF + 1024)         // W_ih (f32[256])
#define WLIN_OFF   (WIH_OFF + 1024)          // W_lin (f32[256])
#define PART_OFF   (WLIN_OFF + 1024)         // o partials (f32[32][8])
#define INB_OFF    (PART_OFF + 1024)         // per-step input (f32[32])
#define OVAL_OFF   (INB_OFF + 128)           // fed-back output (f32[32])
#define SMEM_BYTES (OVAL_OFF + 128)          // 169216 B  (< 320 KB WGP LDS)

// Hardware tanh (CDNA5 TRANS32 op).
__device__ __forceinline__ float fast_tanh(float v) {
#if __has_builtin(__builtin_amdgcn_tanhf)
    return __builtin_amdgcn_tanhf(v);
#elif __has_builtin(__builtin_amdgcn_tanh_f32)
    return __builtin_amdgcn_tanh_f32(v);
#else
    float r;
    asm("v_tanh_f32 %0, %1\n\tv_nop" : "=v"(r) : "v"(v));
    return r;
#endif
}

__global__ __launch_bounds__(NTHREADS)
void rnn_scan_kernel(const float* __restrict__ x,
                     const float* __restrict__ W_ih,
                     const float* __restrict__ b_ih,
                     const float* __restrict__ W_hh,
                     const float* __restrict__ b_hh,
                     const float* __restrict__ W_lin,
                     const float* __restrict__ b_lin,
                     float* __restrict__ out,
                     int L, int T)
{
    extern __shared__ char smem[];
    __bf16* WB    = (__bf16*)(smem + WB_OFF);
    __bf16* hbuf  = (__bf16*)(smem + HBUF_OFF);
    float*  biasb = (float*)(smem + BIAS_OFF);
    float*  wihb  = (float*)(smem + WIH_OFF);
    float*  wlinb = (float*)(smem + WLIN_OFF);
    float*  partb = (float*)(smem + PART_OFF);
    float*  inb   = (float*)(smem + INB_OFF);
    float*  ovalb = (float*)(smem + OVAL_OFF);

    const int tid     = threadIdx.x;
    const int rowbase = blockIdx.x * BT;

    // ---------------- one-time setup ----------------
    biasb[tid] = b_ih[tid] + b_hh[tid];
    wihb[tid]  = W_ih[tid];
    wlinb[tid] = W_lin[tid];

    // W_hh -> LDS, pre-swizzled into WMMA B-matrix (32x16 bf16) fragment layout.
    // B[k][n] = W_hh[coltile*16 + n][k].  Entry (c,kc,lane): 16 contiguous bf16:
    //   lanes 0-15  : n=lane,    k = kc*32 + {0..7, 16..23}
    //   lanes 16-31 : n=lane-16, k = kc*32 + {8..15, 24..31}
    for (int e = tid; e < 16*8*32; e += NTHREADS) {
        const int c  = e >> 8;
        const int kc = (e >> 5) & 7;
        const int l  = e & 31;
        const int row   = c * 16 + (l & 15);
        const int kbase = kc * 32 + ((l >> 4) << 3);
        const float* src = W_hh + row * HDIM + kbase;
        __bf16*      dst = WB + e * 16;
        #pragma unroll
        for (int j = 0; j < 16; ++j) dst[j] = (__bf16)src[j < 8 ? j : j + 8];
    }
    // h0 = 0
    for (int i = tid; i < 2*BT*HST; i += NTHREADS) hbuf[i] = (__bf16)0.f;
    if (tid < BT) ovalb[tid] = 0.f;
    __syncthreads();

    // ---------------- per-wave static assignment ----------------
    const int wave  = tid >> 5;        // 8 waves
    const int lane  = tid & 31;
    const int rtile = wave >> 2;       // 0..1 : which 16-row tile
    const int cbase = (wave & 3) * 4;  // 4 col-tiles per wave (16 total)
    const int lhalf = lane >> 4;       // 0/1
    const int lmod  = lane & 15;

    // constants hoisted to registers
    float biasv[4], wihv[4];
    #pragma unroll
    for (int q = 0; q < 4; ++q) {
        const int col = (cbase + q) * 16 + lmod;
        biasv[q] = biasb[col];
        wihv[q]  = wihb[col];
    }
    const int oro = tid >> 3;          // readout row (0..31)
    const int och = tid & 7;           // readout chunk (32 cols each)
    float wl[32];
    #pragma unroll
    for (int j = 0; j < 32; ++j) wl[j] = wlinb[och * 32 + j];
    const float blin = b_lin[0];

    // software pipeline for x: preload step 0's input before the loop
    float xnext = 0.f;
    if (tid < BT && L > 0) xnext = x[(size_t)(rowbase + tid) * L];

    // ---------------- time scan: L real steps + (T-L) autoregressive ----------------
    int p = 0;
    for (int t = 0; t < T; ++t) {
        // stage A: publish this step's per-row input (x_t or fed-back o)
        if (tid < BT) {
            inb[tid] = (t < L) ? xnext : ovalb[tid];
        }
        __syncthreads();
        // prefetch next step's x; consumed only at next iteration's stage A,
        // so the HBM latency hides behind this step's WMMA work.
        if (tid < BT && (t + 1) < L) {
            xnext = x[(size_t)(rowbase + tid) * L + (t + 1)];
        }

        __bf16* hr = hbuf + p * (BT * HST);        // read h_t
        __bf16* hw = hbuf + (p ^ 1) * (BT * HST);  // write h_{t+1}

        // stage B: H' = tanh(h Whh^T + bias + in*Wih) via bf16 WMMA, f32 accum.
        // A-fragment (16x32 bf16): row = lane%16 of this row-tile,
        //   lanes 0-15: K = kc*32+{0..7, 16..23}; lanes 16-31: K = kc*32+{8..15, 24..31}
        v16bf afrag[8];
        {
            const __bf16* abase = hr + (rtile * 16 + lmod) * HST;
            #pragma unroll
            for (int kc = 0; kc < 8; ++kc) {
                const int k0 = kc * 32 + lhalf * 8;
                union { uint32x4 q[2]; v16bf v; } f;
                f.q[0] = *(const uint32x4*)(abase + k0);
                f.q[1] = *(const uint32x4*)(abase + k0 + 16);
                afrag[kc] = f.v;
            }
        }
        float inrow[8];
        {
            const int rb = rtile * 16 + lhalf * 8;
            #pragma unroll
            for (int r = 0; r < 8; ++r) inrow[r] = inb[rb + r];
        }

        // q-outer: one live accumulator chain (low reg pressure -> deep LDS
        // load-ahead); bias + in*Wih folded into the first WMMA's C operand.
        #pragma unroll
        for (int q = 0; q < 4; ++q) {
            const int c = cbase + q;
            v8f acc;
            #pragma unroll
            for (int r = 0; r < 8; ++r) {
                acc[r] = __builtin_fmaf(inrow[r], wihv[q], biasv[q]);
            }
            #pragma unroll
            for (int kc = 0; kc < 8; ++kc) {
                const __bf16* src = WB + (((c << 3) + kc) * 32 + lane) * 16;
                union { uint32x4 q2[2]; v16bf v; } bfr;
                bfr.q2[0] = *(const uint32x4*)(src);
                bfr.q2[1] = *(const uint32x4*)(src + 8);
                acc = __builtin_amdgcn_wmma_f32_16x16x32_bf16(
                        false, afrag[kc], false, bfr.v, (short)0, acc, false, false);
            }
            // epilogue: C/D layout -> lane n=lmod, rows r + 8*lhalf
            const int col = c * 16 + lmod;
            const int rb  = rtile * 16 + lhalf * 8;
            #pragma unroll
            for (int r = 0; r < 8; ++r) {
                hw[(rb + r) * HST + col] = (__bf16)fast_tanh(acc[r]);
            }
        }
        __syncthreads();

        // stage C: readout partials  o[b] = sum_j h'[b,j] * W_lin[j]
        {
            const __bf16* hrow = hw + oro * HST + och * 32;
            float s = 0.f;
            #pragma unroll
            for (int g = 0; g < 4; ++g) {
                union { uint32x4 q; v8bf b; } u;
                u.q = *(const uint32x4*)(hrow + g * 8);
                #pragma unroll
                for (int j = 0; j < 8; ++j) {
                    s += (float)u.b[j] * wl[g * 8 + j];
                }
            }
            partb[tid] = s;
        }
        __syncthreads();

        // stage D: finalize readout, store, keep for autoregressive feedback
        if (tid < BT) {
            float o = blin;
            #pragma unroll
            for (int j = 0; j < 8; ++j) o += partb[tid * 8 + j];
            ovalb[tid] = o;
            out[(size_t)(rowbase + tid) * T + t] = o;
        }
        p ^= 1;
    }
}

extern "C" void kernel_launch(void* const* d_in, const int* in_sizes, int n_in,
                              void* d_out, int out_size, void* d_ws, size_t ws_size,
                              hipStream_t stream) {
    const float* x     = (const float*)d_in[0];
    const float* W_ih  = (const float*)d_in[1];
    const float* b_ih  = (const float*)d_in[2];
    const float* W_hh  = (const float*)d_in[3];
    const float* b_hh  = (const float*)d_in[4];
    const float* W_lin = (const float*)d_in[5];
    const float* b_lin = (const float*)d_in[6];
    float* out = (float*)d_out;

    const int L = 2048;                 // reference constant
    const int B = in_sizes[0] / L;      // 4096
    const int T = out_size / B;         // L + future (future scalar lives on device)

    hipFuncSetAttribute(reinterpret_cast<const void*>(rnn_scan_kernel),
                        hipFuncAttributeMaxDynamicSharedMemorySize, SMEM_BYTES);

    rnn_scan_kernel<<<dim3(B / BT), dim3(NTHREADS), SMEM_BYTES, stream>>>(
        x, W_ih, b_ih, W_hh, b_hh, W_lin, b_lin, out, L, T);
}
